// MasterModel_16690242912342
// MI455X (gfx1250) — matrix operand order repaired and checked by reading, not compiled, not run on hardware
//
#include <hip/hip_runtime.h>
#include <hip/hip_bf16.h>

// ---------------------------------------------------------------------------
// MasterModel forward for MI455X (gfx1250), wave32 + WMMA bf16.
// All GEMM-shaped stages use v_wmma_f32_16x16x32_bf16 with 2x2 register
// blocking per wave (32x32 output block): each A/B fragment is reused twice,
// halving vmem per WMMA vs a 16x16-per-wave scheme.
//
// Pipeline (all on `stream`, in-order):
//   prep_x        : x[B,1024,30] f32 -> xp[B,32,1024] bf16 (L padded with 0)
//   reorder_conv_w: conv w [Co,Ci,3] f32 -> [Co, 3*Ci] bf16  (K = tap-major)
//   cast_f32_bf16 : fc_w / e_w1 -> bf16 ([N][K] row-major already)
//   conv1_wmma    : implicit GEMM M=61440 K=3072 N=512 -> x1[B,512,30] f32 (relu)
//   pool1         : x1 -> p1p[B,18,512] bf16 (maxpool k2 s2 pad1, L padded 0)
//   conv2_wmma    : implicit GEMM M=32768 K=1536 N=256 -> x2[B,256,16] f32 (relu)
//   pool2         : x2 -> p2b[B,2304] bf16  ((c,j) order == reference reshape)
//   gemm_relu_bf16: fc   M=2048 K=2304 N=1024 -> sf[B,1024] bf16 (relu)
//   gemm_relu_bf16: experts (z=8) M=2048 K=1024 N=512 -> h[8,B,512] bf16 (relu)
//   expert_head   : per-sample select expert, e_w2 matvec + agg -> agg[B,2]
//   fin_kernel    : per-sample 19458-wide dot with fin_w -> out[B,2]
// ---------------------------------------------------------------------------

#define BB 2048

typedef __attribute__((ext_vector_type(16))) __bf16 v16bf;
typedef __attribute__((ext_vector_type(8)))  __bf16 v8bf;
typedef __attribute__((ext_vector_type(8)))  float  v8f;

// A fragment (16x32 bf16): lane L holds row M=L%16; with half=L/16 the 16
// values live at contiguous K = half*16 + [0..15] (matches B fragment).
__device__ __forceinline__ v16bf load_a_frag(const __bf16* __restrict__ row, int half) {
    return *(const v16bf*)(row + half * 16);
}
// B fragment (32x16 bf16): lane L holds column N=L%16, contiguous 16 K values
// starting at half*16. `col` points at W[N][kTile] with W stored [N][K].
__device__ __forceinline__ v16bf load_b_frag(const __bf16* __restrict__ col, int half) {
    return *(const v16bf*)(col + half * 16);
}
__device__ __forceinline__ v8f wmma_bf16(v16bf a, v16bf b, v8f c) {
    return __builtin_amdgcn_wmma_f32_16x16x32_bf16(false, a, false, b, (short)0, c,
                                                   false, false);
}

// ---------------- prep kernels ----------------
__global__ void cast_f32_bf16(const float* __restrict__ src, __bf16* __restrict__ dst, long n) {
    long i = (long)blockIdx.x * blockDim.x + threadIdx.x;
    if (i < n) dst[i] = (__bf16)src[i];
}

// conv w [Co][Ci][3] -> out[Co][3*Ci] with k = t*Ci + c
__global__ void reorder_conv_w(const float* __restrict__ w, __bf16* __restrict__ out,
                               int cin, long n) {
    long i = (long)blockIdx.x * blockDim.x + threadIdx.x;
    if (i >= n) return;
    long K = 3L * cin;
    long nIdx = i / K;
    int  k    = (int)(i % K);
    int  t = k / cin, c = k % cin;
    out[i] = (__bf16)w[(nIdx * cin + c) * 3 + t];
}

// x[B][1024][30] -> xp[B][32][1024], rows 0 and 31 zero (lp = l+1)
__global__ void prep_x(const float* __restrict__ x, __bf16* __restrict__ xp) {
    long i = (long)blockIdx.x * blockDim.x + threadIdx.x;
    if (i >= (long)BB * 32 * 1024) return;
    int  cin = (int)(i & 1023);
    long r   = i >> 10;          // b*32 + lp
    int  lp  = (int)(r & 31);
    long b   = r >> 5;
    float v = 0.f;
    if (lp >= 1 && lp <= 30) v = x[(b * 1024 + cin) * 30 + (lp - 1)];
    xp[i] = (__bf16)v;
}

// ---------------- conv1: M=61440 K=3072 N=512, 32x32 per wave ---------------
__global__ void __launch_bounds__(128)
conv1_wmma(const __bf16* __restrict__ xp, const __bf16* __restrict__ w,
           const float* __restrict__ bias, float* __restrict__ x1) {
    int lane = threadIdx.x & 31, wave = threadIdx.x >> 5;
    int half = lane >> 4, l16 = lane & 15;
    int mBase = blockIdx.x << 5;                       // 32 rows per wave
    int nBase = ((blockIdx.y << 2) + wave) << 5;       // 32 cols per wave
    int mr0 = mBase + l16, mr1 = mBase + 16 + l16;
    int b0 = mr0 / 30, l0 = mr0 % 30;
    int b1 = mr1 / 30, l1 = mr1 % 30;
    const __bf16* aRow0 = xp + ((long)b0 * 32 + l0) * 1024;   // tap t -> +t*1024
    const __bf16* aRow1 = xp + ((long)b1 * 32 + l1) * 1024;
    const __bf16* wRow0 = w + (long)(nBase + l16) * 3072;
    const __bf16* wRow1 = w + (long)(nBase + 16 + l16) * 3072;
    v8f acc00 = {0.f,0.f,0.f,0.f,0.f,0.f,0.f,0.f};
    v8f acc01 = acc00, acc10 = acc00, acc11 = acc00;
    for (int k = 0; k < 3072; k += 32) {
        int t = k >> 10, cb = k & 1023;
        long ao = ((long)t << 10) + cb;
        v16bf a0 = load_a_frag(aRow0 + ao, half);
        v16bf a1 = load_a_frag(aRow1 + ao, half);
        v16bf bm0 = load_b_frag(wRow0 + k, half);
        v16bf bm1 = load_b_frag(wRow1 + k, half);
        acc00 = wmma_bf16(a0, bm0, acc00);
        acc01 = wmma_bf16(a0, bm1, acc01);
        acc10 = wmma_bf16(a1, bm0, acc10);
        acc11 = wmma_bf16(a1, bm1, acc11);
    }
    v8f accs[2][2] = {{acc00, acc01}, {acc10, acc11}};
#pragma unroll
    for (int mi = 0; mi < 2; ++mi)
#pragma unroll
        for (int ni = 0; ni < 2; ++ni) {
            int n = nBase + ni * 16 + l16;
            float bv = bias[n];
#pragma unroll
            for (int v = 0; v < 8; ++v) {
                int m = mBase + mi * 16 + v + (half << 3);
                int bb = m / 30, ll = m % 30;
                x1[((long)bb * 512 + n) * 30 + ll] = fmaxf(accs[mi][ni][v] + bv, 0.f);
            }
        }
}

// x1[B][512][30] -> p1p[B][18][512], rows 0/17 zero (lp = j+1)
__global__ void pool1(const float* __restrict__ x1, __bf16* __restrict__ p1p) {
    long i = (long)blockIdx.x * blockDim.x + threadIdx.x;
    if (i >= (long)BB * 18 * 512) return;
    int  c  = (int)(i & 511);
    long r  = i >> 9;
    int  lp = (int)(r % 18);
    long b  = r / 18;
    float v = 0.f;
    if (lp >= 1 && lp <= 16) {
        int j = lp - 1;
        const float* base = x1 + (b * 512 + c) * 30;
        int i0 = 2 * j - 1, i1 = 2 * j;
        float v0 = (i0 >= 0) ? base[i0] : -__builtin_inff();
        float v1 = (i1 <= 29) ? base[i1] : -__builtin_inff();
        v = fmaxf(v0, v1);
    }
    p1p[i] = (__bf16)v;
}

// ---------------- conv2: M=32768 K=1536 N=256, 32x32 per wave ---------------
__global__ void __launch_bounds__(128)
conv2_wmma(const __bf16* __restrict__ p1p, const __bf16* __restrict__ w,
           const float* __restrict__ bias, float* __restrict__ x2) {
    int lane = threadIdx.x & 31, wave = threadIdx.x >> 5;
    int half = lane >> 4, l16 = lane & 15;
    int mBase = blockIdx.x << 5;
    int nBase = ((blockIdx.y << 2) + wave) << 5;
    int mr0 = mBase + l16, mr1 = mBase + 16 + l16;
    int b0 = mr0 >> 4, j0 = mr0 & 15;
    int b1 = mr1 >> 4, j1 = mr1 & 15;
    const __bf16* aRow0 = p1p + ((long)b0 * 18 + j0) * 512;   // tap t -> +t*512
    const __bf16* aRow1 = p1p + ((long)b1 * 18 + j1) * 512;
    const __bf16* wRow0 = w + (long)(nBase + l16) * 1536;
    const __bf16* wRow1 = w + (long)(nBase + 16 + l16) * 1536;
    v8f acc00 = {0.f,0.f,0.f,0.f,0.f,0.f,0.f,0.f};
    v8f acc01 = acc00, acc10 = acc00, acc11 = acc00;
    for (int k = 0; k < 1536; k += 32) {
        int t = k >> 9, cb = k & 511;
        long ao = ((long)t << 9) + cb;
        v16bf a0 = load_a_frag(aRow0 + ao, half);
        v16bf a1 = load_a_frag(aRow1 + ao, half);
        v16bf bm0 = load_b_frag(wRow0 + k, half);
        v16bf bm1 = load_b_frag(wRow1 + k, half);
        acc00 = wmma_bf16(a0, bm0, acc00);
        acc01 = wmma_bf16(a0, bm1, acc01);
        acc10 = wmma_bf16(a1, bm0, acc10);
        acc11 = wmma_bf16(a1, bm1, acc11);
    }
    v8f accs[2][2] = {{acc00, acc01}, {acc10, acc11}};
#pragma unroll
    for (int mi = 0; mi < 2; ++mi)
#pragma unroll
        for (int ni = 0; ni < 2; ++ni) {
            int n = nBase + ni * 16 + l16;
            float bv = bias[n];
#pragma unroll
            for (int v = 0; v < 8; ++v) {
                int m = mBase + mi * 16 + v + (half << 3);
                int bb = m >> 4, jj = m & 15;
                x2[((long)bb * 256 + n) * 16 + jj] = fmaxf(accs[mi][ni][v] + bv, 0.f);
            }
        }
}

// x2[B][256][16] -> p2b[B][2304] in (c,j) order (== p2.reshape(b,-1))
__global__ void pool2(const float* __restrict__ x2, __bf16* __restrict__ p2b) {
    long i = (long)blockIdx.x * blockDim.x + threadIdx.x;
    if (i >= (long)BB * 2304) return;
    long b = i / 2304;
    int  r = (int)(i % 2304);
    int  c = r / 9, j = r % 9;
    const float* base = x2 + (b * 256 + c) * 16;
    int i0 = 2 * j - 1, i1 = 2 * j;
    float v0 = (i0 >= 0) ? base[i0] : -__builtin_inff();
    float v1 = (i1 <= 15) ? base[i1] : -__builtin_inff();
    p2b[i] = (__bf16)fmaxf(v0, v1);
}

// ---------------- dense GEMM + bias + relu, 32x32 per wave ------------------
__global__ void __launch_bounds__(128)
gemm_relu_bf16(const __bf16* __restrict__ A, const __bf16* __restrict__ W,
               const float* __restrict__ bias, __bf16* __restrict__ D,
               int K, int N, long sA, long sW, long sBias, long sD) {
    int lane = threadIdx.x & 31, wave = threadIdx.x >> 5;
    int half = lane >> 4, l16 = lane & 15;
    long z = blockIdx.z;
    A += z * sA; W += z * sW; bias += z * sBias; D += z * sD;
    int mBase = blockIdx.x << 5;
    int nBase = ((blockIdx.y << 2) + wave) << 5;
    const __bf16* aRow0 = A + (long)(mBase + l16) * K;
    const __bf16* aRow1 = A + (long)(mBase + 16 + l16) * K;
    const __bf16* wRow0 = W + (long)(nBase + l16) * K;
    const __bf16* wRow1 = W + (long)(nBase + 16 + l16) * K;
    v8f acc00 = {0.f,0.f,0.f,0.f,0.f,0.f,0.f,0.f};
    v8f acc01 = acc00, acc10 = acc00, acc11 = acc00;
    for (int k = 0; k < K; k += 32) {
        v16bf a0 = load_a_frag(aRow0 + k, half);
        v16bf a1 = load_a_frag(aRow1 + k, half);
        v16bf bm0 = load_b_frag(wRow0 + k, half);
        v16bf bm1 = load_b_frag(wRow1 + k, half);
        acc00 = wmma_bf16(a0, bm0, acc00);
        acc01 = wmma_bf16(a0, bm1, acc01);
        acc10 = wmma_bf16(a1, bm0, acc10);
        acc11 = wmma_bf16(a1, bm1, acc11);
    }
    v8f accs[2][2] = {{acc00, acc01}, {acc10, acc11}};
#pragma unroll
    for (int mi = 0; mi < 2; ++mi)
#pragma unroll
        for (int ni = 0; ni < 2; ++ni) {
            int n = nBase + ni * 16 + l16;
            float bv = bias[n];
#pragma unroll
            for (int v = 0; v < 8; ++v) {
                int m = mBase + mi * 16 + v + (half << 3);
                D[(long)m * N + n] = (__bf16)fmaxf(accs[mi][ni][v] + bv, 0.f);
            }
        }
}

// ---------------- expert head + aggregation (one wave per sample) -----------
__global__ void expert_head(const __bf16* __restrict__ h, const int* __restrict__ idx,
                            const float* __restrict__ ew2, const float* __restrict__ eb2,
                            const float* __restrict__ aggw, const float* __restrict__ aggb,
                            float* __restrict__ agg) {
    int lane = threadIdx.x & 31, wave = threadIdx.x >> 5;
    int b = blockIdx.x * 8 + wave;
    int e = idx[b];
    const __bf16* hrow = h + ((long)e * BB + b) * 512;
    const float* w0 = ew2 + ((long)e * 2 + 0) * 512;
    const float* w1 = ew2 + ((long)e * 2 + 1) * 512;
    float s0 = 0.f, s1 = 0.f;
    for (int n = lane; n < 512; n += 32) {
        float hv = (float)hrow[n];
        s0 += hv * w0[n];
        s1 += hv * w1[n];
    }
#pragma unroll
    for (int o = 16; o > 0; o >>= 1) {
        s0 += __shfl_down(s0, o, 32);
        s1 += __shfl_down(s1, o, 32);
    }
    if (lane == 0) {
        float sel0 = s0 + eb2[e * 2 + 0];
        float sel1 = s1 + eb2[e * 2 + 1];
        agg[b * 2 + 0] = aggb[0] + sel0 * aggw[e * 2 + 0]      + sel1 * aggw[e * 2 + 1];
        agg[b * 2 + 1] = aggb[1] + sel0 * aggw[16 + e * 2 + 0] + sel1 * aggw[16 + e * 2 + 1];
    }
}

// ---------------- final 19458-wide projection (one block per sample) --------
__global__ void __launch_bounds__(256)
fin_kernel(const float* __restrict__ agg, const float* __restrict__ x1,
           const float* __restrict__ x2, const float* __restrict__ fw,
           const float* __restrict__ fb, float* __restrict__ out) {
    __shared__ float r0[256], r1[256];
    int b = blockIdx.x, t = threadIdx.x;
    const float* x1b = x1 + (long)b * 15360;  // x1.reshape order: co*30 + l
    const float* x2b = x2 + (long)b * 4096;   // x2.reshape order: c*16 + j
    float s0 = 0.f, s1 = 0.f;
    for (int i = t; i < 19458; i += 256) {
        float v = (i < 2) ? agg[b * 2 + i]
                          : (i < 15362 ? x1b[i - 2] : x2b[i - 15362]);
        s0 += v * fw[i];
        s1 += v * fw[19458 + i];
    }
    r0[t] = s0; r1[t] = s1;
    __syncthreads();
    for (int o = 128; o > 0; o >>= 1) {
        if (t < o) { r0[t] += r0[t + o]; r1[t] += r1[t + o]; }
        __syncthreads();
    }
    if (t == 0) {
        out[b * 2 + 0] = r0[0] + fb[0];
        out[b * 2 + 1] = r1[0] + fb[1];
    }
}

// ---------------------------------------------------------------------------
extern "C" void kernel_launch(void* const* d_in, const int* in_sizes, int n_in,
                              void* d_out, int out_size, void* d_ws, size_t ws_size,
                              hipStream_t stream) {
    (void)in_sizes; (void)n_in; (void)out_size; (void)ws_size;
    const float* x    = (const float*)d_in[0];
    const int*   eidx = (const int*)  d_in[1];
    const float* c1w  = (const float*)d_in[2];
    const float* c1b  = (const float*)d_in[3];
    const float* c2w  = (const float*)d_in[4];
    const float* c2b  = (const float*)d_in[5];
    const float* fcw  = (const float*)d_in[6];
    const float* fcb  = (const float*)d_in[7];
    const float* ew1  = (const float*)d_in[8];
    const float* eb1  = (const float*)d_in[9];
    const float* ew2  = (const float*)d_in[10];
    const float* eb2  = (const float*)d_in[11];
    const float* aggw = (const float*)d_in[12];
    const float* aggb = (const float*)d_in[13];
    const float* finw = (const float*)d_in[14];
    const float* finb = (const float*)d_in[15];
    float* out = (float*)d_out;

    char* ws = (char*)d_ws;
    size_t off = 0;
    auto alloc = [&](size_t bytes) -> void* {
        void* p = ws + off;
        off = (off + bytes + 255) & ~(size_t)255;
        return p;
    };
    __bf16* xp   = (__bf16*)alloc((size_t)BB * 32 * 1024 * 2);  // 134 MB
    __bf16* w1b  = (__bf16*)alloc((size_t)512 * 3072 * 2);
    __bf16* w2b  = (__bf16*)alloc((size_t)256 * 1536 * 2);
    __bf16* wfcb = (__bf16*)alloc((size_t)1024 * 2304 * 2);
    __bf16* we1b = (__bf16*)alloc((size_t)8 * 512 * 1024 * 2);
    float*  x1   = (float*) alloc((size_t)BB * 512 * 30 * 4);   // 126 MB
    __bf16* p1p  = (__bf16*)alloc((size_t)BB * 18 * 512 * 2);
    float*  x2   = (float*) alloc((size_t)BB * 256 * 16 * 4);
    __bf16* p2b  = (__bf16*)alloc((size_t)BB * 2304 * 2);
    __bf16* sf   = (__bf16*)alloc((size_t)BB * 1024 * 2);
    __bf16* h    = (__bf16*)alloc((size_t)8 * BB * 512 * 2);
    float*  agg  = (float*) alloc((size_t)BB * 2 * 4);

    auto gs = [](long n, int b) { return dim3((unsigned)((n + b - 1) / b)); };

    // prep
    prep_x        <<<gs((long)BB * 32 * 1024, 256), 256, 0, stream>>>(x, xp);
    reorder_conv_w<<<gs(512L * 3072, 256), 256, 0, stream>>>(c1w, w1b, 1024, 512L * 3072);
    reorder_conv_w<<<gs(256L * 1536, 256), 256, 0, stream>>>(c2w, w2b, 512, 256L * 1536);
    cast_f32_bf16 <<<gs(1024L * 2304, 256), 256, 0, stream>>>(fcw, wfcb, 1024L * 2304);
    cast_f32_bf16 <<<gs(8L * 512 * 1024, 256), 256, 0, stream>>>(ew1, we1b, 8L * 512 * 1024);

    // backbone (32x32 per wave, 4 waves/block along N)
    conv1_wmma<<<dim3(1920, 4, 1), 128, 0, stream>>>(xp, w1b, c1b, x1);
    pool1     <<<gs((long)BB * 18 * 512, 256), 256, 0, stream>>>(x1, p1p);
    conv2_wmma<<<dim3(1024, 2, 1), 128, 0, stream>>>(p1p, w2b, c2b, x2);
    pool2     <<<gs((long)BB * 2304, 256), 256, 0, stream>>>(x2, p2b);

    // fc: M=2048 K=2304 N=1024
    gemm_relu_bf16<<<dim3(64, 8, 1), 128, 0, stream>>>(p2b, wfcb, fcb, sf,
                                                       2304, 1024, 0, 0, 0, 0);
    // experts (dense over z=8): M=2048 K=1024 N=512
    gemm_relu_bf16<<<dim3(64, 4, 8), 128, 0, stream>>>(sf, we1b, eb1, h,
                                                       1024, 512,
                                                       0, 512L * 1024, 512,
                                                       (long)BB * 512);
    // heads
    expert_head<<<dim3(BB / 8), 256, 0, stream>>>(h, eidx, ew2, eb2, aggw, aggb, agg);
    fin_kernel <<<dim3(BB), 256, 0, stream>>>(agg, x1, x2, finw, finb, out);
}
